// LSTMModelGated_9491877724171
// MI455X (gfx1250) — matrix operand order repaired and checked by reading, not compiled, not run on hardware
//
#include <hip/hip_runtime.h>
#include <hip/hip_bf16.h>

// LSTM fused persistent kernel for gfx1250 (MI455X).
// B=4096, T=512, F=25, H=64, G=4H=256. One wave owns a 16-row batch stripe
// for the whole sequence. Per step: z = [x_t|h_t](16x96) @ [Wx;Wh]^T(96x256)
// as 48 x v_wmma_f32_16x16x32_bf16 (f32 accumulate), then gates in f32.

#define TSTEPS 512
#define FDIM   25
#define HDIM   64
#define GDIM   256          // 4*H
#define KTOT   96           // 32 (F padded) + 64 (H)
#define ROWS_PER_WAVE  16
#define WAVES_PER_BLOCK 2
#define ROWS_PER_BLOCK (ROWS_PER_WAVE * WAVES_PER_BLOCK)
#define BATCH  4096

typedef __attribute__((ext_vector_type(16))) __bf16 v16bf;
typedef __attribute__((ext_vector_type(8)))  __bf16 v8bf;
typedef __attribute__((ext_vector_type(8)))  float  v8f;

__device__ __forceinline__ v16bf cat8(v8bf lo, v8bf hi) {
  return __builtin_shufflevector(lo, hi, 0,1,2,3,4,5,6,7,8,9,10,11,12,13,14,15);
}
__device__ __forceinline__ v8f splat8(float v) {
  v8f r = {v, v, v, v, v, v, v, v};
  return r;
}
// sigmoid / tanh via v_exp_f32 + v_rcp_f32 (TRANS ops co-execute with WMMA)
__device__ __forceinline__ float sigm(float x) {
  return __builtin_amdgcn_rcpf(1.0f + __expf(-x));
}
__device__ __forceinline__ float tanh_fast(float x) {
  return 2.0f * sigm(2.0f * x) - 1.0f;
}

__global__ __launch_bounds__(WAVES_PER_BLOCK * 32, 1)
void lstm_fused_gfx1250(const float* __restrict__ x,
                        const float* __restrict__ Wx_w, const float* __restrict__ Wx_b,
                        const float* __restrict__ Wh_w, const float* __restrict__ Wh_b,
                        const float* __restrict__ fc_w, const float* __restrict__ fc_b,
                        float* __restrict__ out) {
  // Combined weight matrix W^T[g][k], K-contiguous per output column so a
  // B-fragment (32x16) is two ds_load_b128 per lane. k<25 -> Wx, 25..31 -> 0
  // (pad), 32..95 -> Wh. bf16 storage: 256*96*2 = 48 KB.
  __shared__ __align__(16) __bf16 s_wT[GDIM * KTOT];
  __shared__ __align__(16) float  s_bias[GDIM];          // Wx_b + Wh_b
  __shared__ __align__(16) float  s_fcw[HDIM];
  // Wave-private A staging tile: [row][ x_t(32) | h_t(64) ] bf16.
  __shared__ __align__(16) __bf16 s_xh[WAVES_PER_BLOCK][ROWS_PER_WAVE * KTOT];

  const int tid  = threadIdx.x;
  const int lane = tid & 31;
  const int wv   = tid >> 5;
  const int m    = lane & 15;        // row within A tile / column within B tile
  const int hi16 = lane >> 4;        // upper-half-of-wave selector

  // ---- one-time weight preload into LDS (bf16) ----
  for (int idx = tid; idx < GDIM * KTOT; idx += WAVES_PER_BLOCK * 32) {
    int g = idx / KTOT, k = idx - g * KTOT;
    float v = 0.0f;
    if (k < FDIM)      v = Wx_w[g * FDIM + k];
    else if (k >= 32)  v = Wh_w[g * HDIM + (k - 32)];
    s_wT[idx] = (__bf16)v;
  }
  for (int idx = tid; idx < GDIM; idx += WAVES_PER_BLOCK * 32)
    s_bias[idx] = Wx_b[idx] + Wh_b[idx];
  if (tid < HDIM) s_fcw[tid] = fc_w[tid];

  const int grow = blockIdx.x * ROWS_PER_BLOCK + wv * ROWS_PER_WAVE;

  // ---- init wave tile: x_0 in cols [0,32), h_0 = 0 in cols [32,96) ----
  {
    int r  = lane >> 1;
    int ch = (lane & 1) * 16;
    const float* xr = x + ((size_t)(grow + r) * TSTEPS + 0) * FDIM;
    __bf16* dst = &s_xh[wv][r * KTOT + ch];
    #pragma unroll
    for (int c = 0; c < 16; ++c) {
      float v = (ch + c < FDIM) ? xr[ch + c] : 0.0f;
      dst[c] = (__bf16)v;
    }
    __bf16* hz = &s_xh[wv][r * KTOT + 32 + (lane & 1) * 32];
    #pragma unroll
    for (int c = 0; c < 32; ++c) hz[c] = (__bf16)0.0f;
  }
  __syncthreads();   // only barrier: weights + initial tiles visible

  v8f cst[4], hT[4];                 // cell / hidden state, f32 C/D layout
  #pragma unroll
  for (int j = 0; j < 4; ++j) { cst[j] = splat8(0.0f); hT[j] = splat8(0.0f); }

  const __bf16* arow = &s_xh[wv][m * KTOT];
  const int pr = lane >> 1, pch = (lane & 1) * 16;   // x staging role of lane

  #pragma unroll 1
  for (int t = 0; t < TSTEPS; ++t) {
    // Prefetch next timestep's x into registers; latency hidden by WMMAs.
    float xv[16];
    if (t + 1 < TSTEPS) {
      const float* xr = x + ((size_t)(grow + pr) * TSTEPS + (t + 1)) * FDIM;
      #pragma unroll
      for (int c = 0; c < 16; ++c)
        xv[c] = (pch + c < FDIM) ? xr[pch + c] : 0.0f;
      if (t + 2 < TSTEPS)   // global_prefetch_b8 two steps ahead
        __builtin_prefetch(x + ((size_t)(grow + pr) * TSTEPS + (t + 2)) * FDIM, 0, 1);
    }

    // A fragments (16x32 bf16): lane<16 holds K {0..7,16..23}, lane>=16 {8..15,24..31}
    v16bf A[3];
    #pragma unroll
    for (int kc = 0; kc < 3; ++kc) {
      int kb = kc * 32 + hi16 * 8;
      v8bf lo = *(const v8bf*)&arow[kb];
      v8bf hi = *(const v8bf*)&arow[kb + 16];
      A[kc] = cat8(lo, hi);
    }

    // z = bias + [x|h] @ W^T : 16 N-tiles x 3 K-chunks = 48 WMMAs
    v8f z[16];
    #pragma unroll
    for (int nt = 0; nt < 16; ++nt) {
      const int n = nt * 16 + m;
      z[nt] = splat8(s_bias[n]);
      const __bf16* wcol = &s_wT[n * KTOT];
      #pragma unroll
      for (int kc = 0; kc < 3; ++kc) {
        int kb = kc * 32 + hi16 * 8;    // B 32x16: lane<16 K{0..7,16..23}, lane>=16 K{8..15,24..31}
        v8bf b0 = *(const v8bf*)&wcol[kb];
        v8bf b1 = *(const v8bf*)&wcol[kb + 16];
        v16bf Bf = cat8(b0, b1);
        z[nt] = __builtin_amdgcn_wmma_f32_16x16x32_bf16(
                    false, A[kc], false, Bf, (short)0, z[nt], false, false);
      }
    }

    // Gates: N-tiles 0..3 = i, 4..7 = f, 8..11 = g, 12..15 = o (f32, C layout)
    #pragma unroll
    for (int j = 0; j < 4; ++j) {
      #pragma unroll
      for (int e = 0; e < 8; ++e) {
        float iv = sigm(z[j][e]);
        float fv = sigm(z[4 + j][e]);
        float gv = tanh_fast(z[8 + j][e]);
        float ov = sigm(z[12 + j][e]);
        float cv = fv * cst[j][e] + iv * gv;
        cst[j][e] = cv;
        hT[j][e]  = ov * tanh_fast(cv);
      }
    }

    // Recycle h_{t+1} (bf16) into the wave-private A tile; C/D layout:
    // element (VGPR e, lane) = (M = e + hi16*8, N = j*16 + m). LDS is
    // in-order within a wave -> no barrier needed in the time loop.
    #pragma unroll
    for (int j = 0; j < 4; ++j) {
      int nn = 32 + j * 16 + m;
      #pragma unroll
      for (int e = 0; e < 8; ++e)
        s_xh[wv][(e + hi16 * 8) * KTOT + nn] = (__bf16)hT[j][e];
    }
    // Stage x_{t+1}
    if (t + 1 < TSTEPS) {
      __bf16* dst = &s_xh[wv][pr * KTOT + pch];
      #pragma unroll
      for (int c = 0; c < 16; ++c) dst[c] = (__bf16)xv[c];
    }
  }

  // ---- logits = h_T @ fc_w + fc_b, from f32 register tiles ----
  #pragma unroll
  for (int e = 0; e < 8; ++e) {
    float s = 0.0f;
    #pragma unroll
    for (int j = 0; j < 4; ++j) s += hT[j][e] * s_fcw[j * 16 + m];
    // reduce over the 16 lanes of this half-wave (xor masks stay within half)
    s += __shfl_xor(s, 1, 32);
    s += __shfl_xor(s, 2, 32);
    s += __shfl_xor(s, 4, 32);
    s += __shfl_xor(s, 8, 32);
    if (m == 0) out[grow + hi16 * 8 + e] = s + fc_b[0];
  }
}

extern "C" void kernel_launch(void* const* d_in, const int* in_sizes, int n_in,
                              void* d_out, int out_size, void* d_ws, size_t ws_size,
                              hipStream_t stream) {
  const float* x    = (const float*)d_in[0];
  // d_in[1] = mask: all-ones, reference uses last timestep -> unused
  const float* Wx_w = (const float*)d_in[2];
  const float* Wx_b = (const float*)d_in[3];
  const float* Wh_w = (const float*)d_in[4];
  const float* Wh_b = (const float*)d_in[5];
  const float* fc_w = (const float*)d_in[6];
  const float* fc_b = (const float*)d_in[7];
  float* out = (float*)d_out;

  dim3 grid(BATCH / ROWS_PER_BLOCK);       // 128 blocks
  dim3 block(WAVES_PER_BLOCK * 32);        // 64 threads = 2 waves
  lstm_fused_gfx1250<<<grid, block, 0, stream>>>(
      x, Wx_w, Wx_b, Wh_w, Wh_b, fc_w, fc_b, out);
}
